// Model_63350767616210
// MI455X (gfx1250) — matrix operand order repaired and checked by reading, not compiled, not run on hardware
//
#include <hip/hip_runtime.h>
#include <hip/hip_bf16.h>
#include <cmath>

// ---------------------------------------------------------------------------
// MI455X (gfx1250) implementation. All matmuls routed through
// v_wmma_f32_16x16x32_bf16 (bf16 inputs, f32 accumulate). Weights converted to
// bf16 once per call; activations kept in f32 with bf16 shadow copies where
// they feed a GEMM. Sequential scans (tree depth, LSTM time) are minimized by
// precomputing every batchable GEMM up front. Wave tile = 32x64 output
// (2 A-frags x 4 B-frags -> 8 WMMA per 12 b128 loads per k-step). K is
// hardcoded to 512 so all fragment strides are immediate offsets: only two
// per-lane address registers survive in the inner loop.
// ---------------------------------------------------------------------------

typedef __bf16 bfloat;
typedef __attribute__((ext_vector_type(16))) __bf16 v16bf;
typedef __attribute__((ext_vector_type(8)))  __bf16 v8bf;
typedef __attribute__((ext_vector_type(8)))  float  v8f;

union AFrag { v16bf v; v8bf h[2]; };

#define DEPTH 8
#define NNODE 128
#define KCH   16
#define DIM   512
#define BB    128
#define TT    64
#define GK    512   // K dimension of every GEMM in this model (compile-time)

__device__ __forceinline__ float sigmf(float x) { return 1.0f / (1.0f + __expf(-x)); }

// ---------------- small utility kernels ----------------

__global__ __launch_bounds__(256) void k_conv_bf16(const float* __restrict__ s,
                                                   bfloat* __restrict__ d, int n) {
  int i = blockIdx.x * blockDim.x + threadIdx.x;
  if (i < n) d[i] = (bfloat)s[i];
}

__global__ __launch_bounds__(256) void k_add_vec(const float* __restrict__ a,
                                                 const float* __restrict__ b,
                                                 float* __restrict__ o, int n) {
  int i = blockIdx.x * blockDim.x + threadIdx.x;
  if (i < n) o[i] = a[i] + b[i];
}

// gather embedding rows (table is [V, DIM] f32) -> bf16 rows
__global__ __launch_bounds__(256) void k_gather_bf16(const float* __restrict__ table,
                                                     const int* __restrict__ idx,
                                                     bfloat* __restrict__ out, int rows) {
  int r = blockIdx.x;
  if (r >= rows) return;
  const float* src = table + (size_t)idx[r] * DIM;
  bfloat* dst = out + (size_t)r * DIM;
  for (int j = threadIdx.x; j < DIM; j += blockDim.x) dst[j] = (bfloat)src[j];
}

// ---------------- WMMA GEMM: C[M,N] = A[M,GK] @ B[N,GK]^T (+bias[N]) (+addend) --
// One wave computes a 32x64 output tile: 2 A-frags x 4 B-frags -> 8 WMMA per
// 32-wide k-step. Fragment loads use one contiguous K order (not the ISA table order) for A (16x32) and B (32x16)
// fragments: lanes 0-15 / 16-31 split the K range; all loads K-contiguous.
// Inter-fragment strides (16*GK, 32*GK, 48*GK elements) are compile-time, so
// they lower to global_load_b128 immediate offsets off two base registers.
// Requires M % 32 == 0, N % 64 == 0.

__global__ __launch_bounds__(256)
void k_gemm_bf16(const bfloat* __restrict__ A, const bfloat* __restrict__ Bw,
                 const float* __restrict__ bias,
                 const float* __restrict__ addend, int addStride,
                 float* __restrict__ C, bfloat* __restrict__ Cb,
                 int M, int N) {
  const int nT = N >> 6;                     // 64-wide N tiles
  const int mT = M >> 5;                     // 32-wide M tiles
  int wave = blockIdx.x * (blockDim.x >> 5) + (threadIdx.x >> 5);
  if (wave >= mT * nT) return;               // wave-uniform: EXEC stays all-ones
  const int mBase = (wave / nT) * 32;
  const int nBase = (wave % nT) * 64;
  const int lane = threadIdx.x & 31;
  const int half = lane >> 4;                // K-half selector (contiguous K order)
  const int l15  = lane & 15;

  v8f acc00 = {}, acc01 = {}, acc02 = {}, acc03 = {};
  v8f acc10 = {}, acc11 = {}, acc12 = {}, acc13 = {};

  // Two per-lane base pointers; everything else is an immediate offset.
  const bfloat* ap = A  + (size_t)(mBase + l15) * GK + half * 16;
  const bfloat* bp = Bw + (size_t)(nBase + l15) * GK + half * 16;

  #pragma unroll 2
  for (int k0 = 0; k0 < GK; k0 += 32) {
    AFrag a0, a1;
    // A 16x32: lanes0-15 elems0-15 = K[k0..k0+15];
    //          lanes16-31 shifted by 16 (folded into ap via half*16).
    a0.h[0] = *(const v8bf*)(ap);
    a0.h[1] = *(const v8bf*)(ap + 8);
    a1.h[0] = *(const v8bf*)(ap + 16 * GK);
    a1.h[1] = *(const v8bf*)(ap + 16 * GK + 8);
    // B 32x16: lane holds one N column, 16 contiguous K values (half*16 in bp)
    v16bf b0 = *(const v16bf*)(bp);
    v16bf b1 = *(const v16bf*)(bp + 16 * GK);
    v16bf b2 = *(const v16bf*)(bp + 32 * GK);
    v16bf b3 = *(const v16bf*)(bp + 48 * GK);
    acc00 = __builtin_amdgcn_wmma_f32_16x16x32_bf16(false, a0.v, false, b0, (short)0, acc00, false, false);
    acc01 = __builtin_amdgcn_wmma_f32_16x16x32_bf16(false, a0.v, false, b1, (short)0, acc01, false, false);
    acc02 = __builtin_amdgcn_wmma_f32_16x16x32_bf16(false, a0.v, false, b2, (short)0, acc02, false, false);
    acc03 = __builtin_amdgcn_wmma_f32_16x16x32_bf16(false, a0.v, false, b3, (short)0, acc03, false, false);
    acc10 = __builtin_amdgcn_wmma_f32_16x16x32_bf16(false, a1.v, false, b0, (short)0, acc10, false, false);
    acc11 = __builtin_amdgcn_wmma_f32_16x16x32_bf16(false, a1.v, false, b1, (short)0, acc11, false, false);
    acc12 = __builtin_amdgcn_wmma_f32_16x16x32_bf16(false, a1.v, false, b2, (short)0, acc12, false, false);
    acc13 = __builtin_amdgcn_wmma_f32_16x16x32_bf16(false, a1.v, false, b3, (short)0, acc13, false, false);
    ap += 32;
    bp += 32;
  }

  // C/D layout: lane(0-15)->N=lane, VGPR r -> M=r; lane(16-31)->N=lane-16, M=8+r
  v8f accs[2][4] = {{acc00, acc01, acc02, acc03}, {acc10, acc11, acc12, acc13}};
  #pragma unroll
  for (int i = 0; i < 2; i++) {
    #pragma unroll
    for (int j = 0; j < 4; j++) {
      int col = nBase + j * 16 + l15;
      float bi = bias ? bias[col] : 0.0f;
      #pragma unroll
      for (int r = 0; r < 8; r++) {
        int row = mBase + i * 16 + half * 8 + r;
        float v = accs[i][j][r] + bi;
        if (addend) v += addend[(size_t)row * addStride + col];
        size_t o = (size_t)row * N + col;
        if (C)  C[o]  = v;
        if (Cb) Cb[o] = (bfloat)v;
      }
    }
  }
}

// ---------------- tree-LSTM kernels ----------------

// Gather children states for one depth. Reference semantics: child index v in
// [-1,127]; v==-1 masked, v==0 hits the zero row, v>=1 -> prev node v-1.
// Both v<=0 cases contribute exact zeros, so fold them together.
__global__ __launch_bounds__(256)
void k_tree_gather(const int* __restrict__ tidx, const bfloat* __restrict__ h_prev,
                   const float* __restrict__ c_prev,
                   bfloat* __restrict__ h_gath, float* __restrict__ c_gath,
                   bfloat* __restrict__ h_sum) {
  int n = blockIdx.x;
  int idxs[KCH];
  #pragma unroll
  for (int k = 0; k < KCH; k++) idxs[k] = tidx[n * KCH + k];
  for (int j = threadIdx.x; j < DIM; j += blockDim.x) {
    float hs = 0.0f;
    #pragma unroll
    for (int k = 0; k < KCH; k++) {
      float hv = 0.0f, cv = 0.0f;
      int v = idxs[k];
      if (v >= 1) {
        hv = (float)h_prev[(size_t)(v - 1) * DIM + j];
        cv = c_prev[(size_t)(v - 1) * DIM + j];
      }
      h_gath[(size_t)(n * KCH + k) * DIM + j] = (bfloat)hv;
      c_gath[(size_t)(n * KCH + k) * DIM + j] = cv;
      hs += hv;
    }
    h_sum[(size_t)n * DIM + j] = (bfloat)hs;
  }
}

// Gate math for one depth. Wx gate order (reference): f, i, u, o. iuo: i, u, o.
// fk_lin already includes b_f via GEMM bias; iuo includes b_iuo.
__global__ __launch_bounds__(256)
void k_tree_combine(const float* __restrict__ Wx, const float* __restrict__ fk_lin,
                    const float* __restrict__ c_gath, const float* __restrict__ iuo,
                    float* __restrict__ c_state, bfloat* __restrict__ h_state,
                    const float* __restrict__ emb_last,
                    float* __restrict__ code_f32, bfloat* __restrict__ code_bf) {
  int n = blockIdx.x;
  const float* wxr = Wx  + (size_t)n * (4 * DIM);
  const float* iur = iuo + (size_t)n * (3 * DIM);
  for (int j = threadIdx.x; j < DIM; j += blockDim.x) {
    float Wf = wxr[j], Wi = wxr[DIM + j], Wu = wxr[2 * DIM + j], Wo = wxr[3 * DIM + j];
    float facc = 0.0f;
    #pragma unroll
    for (int k = 0; k < KCH; k++) {
      size_t o = (size_t)(n * KCH + k) * DIM + j;
      facc += sigmf(Wf + fk_lin[o]) * c_gath[o];   // masked children have c==0
    }
    float ig = sigmf(iur[j] + Wi);
    float ug = tanhf(iur[DIM + j] + Wu);
    float og = sigmf(iur[2 * DIM + j] + Wo);
    float nc = ig * ug + facc;
    float nh = og * tanhf(nc);
    size_t o = (size_t)n * DIM + j;
    c_state[o] = nc;
    h_state[o] = (bfloat)nh;
    if (emb_last) {                                // last depth: code_vec = h + emb
      float cv = nh + emb_last[o];
      code_f32[o] = cv;
      code_bf[o]  = (bfloat)cv;
    }
  }
}

// ---------------- BiLSTM kernels ----------------

// G[b, 0:2048] = full preactivation (x-proj + bias already added via GEMM
// addend/bias). torch gate order: i, f, g, o.
__global__ __launch_bounds__(256)
void k_lstm_step(const float* __restrict__ G, float* __restrict__ c,
                 bfloat* __restrict__ h_bf, float* __restrict__ h_out) {
  int b = blockIdx.x;
  const float* g = G + (size_t)b * (4 * DIM);
  for (int j = threadIdx.x; j < DIM; j += blockDim.x) {
    float ig = sigmf(g[j]);
    float fg = sigmf(g[DIM + j]);
    float gg = tanhf(g[2 * DIM + j]);
    float og = sigmf(g[3 * DIM + j]);
    size_t o = (size_t)b * DIM + j;
    float nc = fg * c[o] + ig * gg;
    float nh = og * tanhf(nc);
    c[o] = nc;
    h_bf[o] = (bfloat)nh;
    if (h_out) h_out[o] = nh;
  }
}

// nl_vec = h_fwd + h_bwd (f32 straight into d_out, bf16 shadow for the score GEMM)
__global__ __launch_bounds__(256)
void k_nlvec(const float* __restrict__ a, const float* __restrict__ b,
             float* __restrict__ of, bfloat* __restrict__ ob, int n) {
  int i = blockIdx.x * blockDim.x + threadIdx.x;
  if (i < n) { float v = a[i] + b[i]; of[i] = v; ob[i] = (bfloat)v; }
}

// InfoNCE loss: one 128-thread block, one row per thread, LDS tree-reduce.
__global__ __launch_bounds__(128)
void k_loss(const float* __restrict__ scores, float* __restrict__ out) {
  __shared__ float red[BB];
  int i = threadIdx.x;
  const float* row = scores + (size_t)i * BB;
  float mx = -3.4e38f;
  for (int j = 0; j < BB; j++) mx = fmaxf(mx, row[j]);
  float s = 0.0f;
  for (int j = 0; j < BB; j++) s += __expf(row[j] - mx);
  red[i] = row[i] - mx - __logf(s);
  __syncthreads();
  for (int st = BB / 2; st > 0; st >>= 1) {
    if (i < st) red[i] += red[i + st];
    __syncthreads();
  }
  if (i == 0) out[0] = -red[0] / (float)BB;
}

// ---------------------------------------------------------------------------

extern "C" void kernel_launch(void* const* d_in, const int* in_sizes, int n_in,
                              void* d_out, int out_size, void* d_ws, size_t ws_size,
                              hipStream_t stream) {
  (void)in_sizes; (void)n_in; (void)out_size; (void)ws_size;

  const int*   x_tree    = (const int*)d_in[0];   // [8,128,1]
  const int*   tree_idx  = (const int*)d_in[1];   // [8,128,16]
  const int*   nl_inputs = (const int*)d_in[2];   // [128,64]
  const float* E_tree = (const float*)d_in[3];
  const float* W_lin  = (const float*)d_in[4];
  const float* b_lin  = (const float*)d_in[5];
  const float* W_t    = (const float*)d_in[6];
  const float* b_t    = (const float*)d_in[7];
  const float* U_f    = (const float*)d_in[8];
  const float* b_f    = (const float*)d_in[9];
  const float* U_iuo  = (const float*)d_in[10];
  const float* b_iuo  = (const float*)d_in[11];
  const float* Emb    = (const float*)d_in[12];
  const float* Wih_f  = (const float*)d_in[13];
  const float* Whh_f  = (const float*)d_in[14];
  const float* bih_f  = (const float*)d_in[15];
  const float* bhh_f  = (const float*)d_in[16];
  const float* Wih_b  = (const float*)d_in[17];
  const float* Whh_b  = (const float*)d_in[18];
  const float* bih_b  = (const float*)d_in[19];
  const float* bhh_b  = (const float*)d_in[20];

  float* out = (float*)d_out;
  float* out_loss = out;                       // [1]
  float* out_code = out + 1;                   // [128*512]
  float* out_nl   = out + 1 + BB * DIM;        // [128*512]

  // ---- workspace carve (256B aligned) ----
  char* base = (char*)d_ws;
  size_t off = 0;
  auto carve = [&](size_t bytes) -> void* {
    void* p = base + off;
    off += (bytes + 255) & ~(size_t)255;
    return p;
  };
  bfloat* Wlin_bf  = (bfloat*)carve((size_t)DIM * DIM * 2);
  bfloat* Wt_bf    = (bfloat*)carve((size_t)4 * DIM * DIM * 2);
  bfloat* Uf_bf    = (bfloat*)carve((size_t)DIM * DIM * 2);
  bfloat* Uiuo_bf  = (bfloat*)carve((size_t)3 * DIM * DIM * 2);
  bfloat* Wihf_bf  = (bfloat*)carve((size_t)4 * DIM * DIM * 2);
  bfloat* Whhf_bf  = (bfloat*)carve((size_t)4 * DIM * DIM * 2);
  bfloat* Wihb_bf  = (bfloat*)carve((size_t)4 * DIM * DIM * 2);
  bfloat* Whhb_bf  = (bfloat*)carve((size_t)4 * DIM * DIM * 2);
  float*  biasF    = (float*)carve(4 * DIM * 4);      // bih_f + bhh_f
  float*  biasBk   = (float*)carve(4 * DIM * 4);      // bih_b + bhh_b
  bfloat* xin_bf   = (bfloat*)carve((size_t)DEPTH * NNODE * DIM * 2);
  float*  emb_f32  = (float*)carve((size_t)DEPTH * NNODE * DIM * 4);
  bfloat* emb_bf   = (bfloat*)carve((size_t)DEPTH * NNODE * DIM * 2);
  float*  Wx_all   = (float*)carve((size_t)DEPTH * NNODE * 4 * DIM * 4);
  bfloat* h_gath   = (bfloat*)carve((size_t)NNODE * KCH * DIM * 2);
  float*  c_gath   = (float*)carve((size_t)NNODE * KCH * DIM * 4);
  bfloat* h_sum    = (bfloat*)carve((size_t)NNODE * DIM * 2);
  float*  fk_lin   = (float*)carve((size_t)NNODE * KCH * DIM * 4);
  float*  iuo_f    = (float*)carve((size_t)NNODE * 3 * DIM * 4);
  bfloat* h_state  = (bfloat*)carve((size_t)NNODE * DIM * 2);
  float*  c_state  = (float*)carve((size_t)NNODE * DIM * 4);
  bfloat* code_bf  = (bfloat*)carve((size_t)BB * DIM * 2);
  bfloat* nl_bf    = (bfloat*)carve((size_t)BB * TT * DIM * 2);
  float*  Xproj    = (float*)carve((size_t)BB * TT * 4 * DIM * 4);   // 64 MB, reused per dir
  float*  G        = (float*)carve((size_t)BB * 4 * DIM * 4);
  bfloat* hl_bf    = (bfloat*)carve((size_t)BB * DIM * 2);
  float*  cl       = (float*)carve((size_t)BB * DIM * 4);
  float*  h_fwd    = (float*)carve((size_t)BB * DIM * 4);
  float*  h_bwd    = (float*)carve((size_t)BB * DIM * 4);
  bfloat* nlvec_bf = (bfloat*)carve((size_t)BB * DIM * 2);
  float*  scores   = (float*)carve((size_t)BB * BB * 4);

  auto conv = [&](const float* s, bfloat* d, int n) {
    k_conv_bf16<<<(n + 255) / 256, 256, 0, stream>>>(s, d, n);
  };
  auto gemm = [&](const bfloat* A, const bfloat* B, const float* bias,
                  const float* add, int addStride, float* C, bfloat* Cb,
                  int M, int N) {
    int waves = (M / 32) * (N / 64);
    k_gemm_bf16<<<(waves + 7) / 8, 256, 0, stream>>>(A, B, bias, add, addStride,
                                                     C, Cb, M, N);
  };

  // ---- weight conversion + bias folding ----
  conv(W_lin, Wlin_bf, DIM * DIM);
  conv(W_t,   Wt_bf,   4 * DIM * DIM);
  conv(U_f,   Uf_bf,   DIM * DIM);
  conv(U_iuo, Uiuo_bf, 3 * DIM * DIM);
  conv(Wih_f, Wihf_bf, 4 * DIM * DIM);
  conv(Whh_f, Whhf_bf, 4 * DIM * DIM);
  conv(Wih_b, Wihb_bf, 4 * DIM * DIM);
  conv(Whh_b, Whhb_bf, 4 * DIM * DIM);
  k_add_vec<<<(4 * DIM + 255) / 256, 256, 0, stream>>>(bih_f, bhh_f, biasF, 4 * DIM);
  k_add_vec<<<(4 * DIM + 255) / 256, 256, 0, stream>>>(bih_b, bhh_b, biasBk, 4 * DIM);

  // ---- embeddings (gather + convert), batched tree GEMMs ----
  k_gather_bf16<<<DEPTH * NNODE, 256, 0, stream>>>(E_tree, x_tree, xin_bf, DEPTH * NNODE);
  k_gather_bf16<<<BB * TT, 256, 0, stream>>>(Emb, nl_inputs, nl_bf, BB * TT);
  // emb = x @ W_lin.T + b_lin   [1024,512]
  gemm(xin_bf, Wlin_bf, b_lin, nullptr, 0, emb_f32, emb_bf, DEPTH * NNODE, DIM);
  // Wx (all depths) = emb @ W_t.T + b_t   [1024,2048]
  gemm(emb_bf, Wt_bf, b_t, nullptr, 0, Wx_all, nullptr, DEPTH * NNODE, 4 * DIM);

  // ---- child-sum tree-LSTM scan over depth ----
  for (int d = 0; d < DEPTH; d++) {
    k_tree_gather<<<NNODE, 256, 0, stream>>>(tree_idx + (size_t)d * NNODE * KCH,
                                             h_state, c_state, h_gath, c_gath, h_sum);
    // fk preact: [2048,512] = h_gath @ U_f.T + b_f
    gemm(h_gath, Uf_bf, b_f, nullptr, 0, fk_lin, nullptr, NNODE * KCH, DIM);
    // iuo: [128,1536] = h_sum @ U_iuo.T + b_iuo
    gemm(h_sum, Uiuo_bf, b_iuo, nullptr, 0, iuo_f, nullptr, NNODE, 3 * DIM);
    bool last = (d == DEPTH - 1);
    k_tree_combine<<<NNODE, 256, 0, stream>>>(
        Wx_all + (size_t)d * NNODE * 4 * DIM, fk_lin, c_gath, iuo_f,
        c_state, h_state,
        last ? emb_f32 + (size_t)(DEPTH - 1) * NNODE * DIM : nullptr,
        out_code, code_bf);
  }

  // ---- BiLSTM forward direction ----
  gemm(nl_bf, Wihf_bf, biasF, nullptr, 0, Xproj, nullptr, BB * TT, 4 * DIM);
  hipMemsetAsync(hl_bf, 0, (size_t)BB * DIM * sizeof(bfloat), stream);
  hipMemsetAsync(cl,    0, (size_t)BB * DIM * sizeof(float),  stream);
  for (int t = 0; t < TT; t++) {
    gemm(hl_bf, Whhf_bf, nullptr, Xproj + (size_t)t * 4 * DIM, TT * 4 * DIM,
         G, nullptr, BB, 4 * DIM);
    k_lstm_step<<<BB, 256, 0, stream>>>(G, cl, hl_bf, (t == TT - 1) ? h_fwd : nullptr);
  }
  // ---- BiLSTM backward direction (reversed sequence; reuse Xproj buffer) ----
  gemm(nl_bf, Wihb_bf, biasBk, nullptr, 0, Xproj, nullptr, BB * TT, 4 * DIM);
  hipMemsetAsync(hl_bf, 0, (size_t)BB * DIM * sizeof(bfloat), stream);
  hipMemsetAsync(cl,    0, (size_t)BB * DIM * sizeof(float),  stream);
  for (int s = 0; s < TT; s++) {
    int t = TT - 1 - s;
    gemm(hl_bf, Whhb_bf, nullptr, Xproj + (size_t)t * 4 * DIM, TT * 4 * DIM,
         G, nullptr, BB, 4 * DIM);
    k_lstm_step<<<BB, 256, 0, stream>>>(G, cl, hl_bf, (s == TT - 1) ? h_bwd : nullptr);
  }

  // ---- nl_vec, scores, loss ----
  k_nlvec<<<(BB * DIM + 255) / 256, 256, 0, stream>>>(h_fwd, h_bwd, out_nl, nlvec_bf, BB * DIM);
  // scores = nl_vec @ code_vec.T   [128,128]
  gemm(nlvec_bf, code_bf, nullptr, nullptr, 0, scores, nullptr, BB, BB);
  k_loss<<<1, BB, 0, stream>>>(scores, out_loss);
}
